// DST_PredRNN_22539988369577
// MI455X (gfx1250) — compile-verified, matrix operand order checked by reading, not run on hardware
//
#include <hip/hip_runtime.h>
#include <hip/hip_bf16.h>

typedef __attribute__((ext_vector_type(16))) _Float16 v16h;
typedef __attribute__((ext_vector_type(8)))  float    v8f;
typedef __attribute__((ext_vector_type(4)))  unsigned int u32x4;

#define NLAY  4
#define HCH   64
#define FCH   16
#define HWD   32
#define TT    20
#define TIN   10
#define BB    16
#define NN    16384   /* B*HW*HW */
#define NSTEP 19

__device__ __forceinline__ float sigmoidf_(float x) { return 1.0f / (1.0f + expf(-x)); }

/* ---------------- weight fp32 -> f16 matrix [M][Kpad], zero padded ---------------- */
__global__ void k_convert_w(const float* __restrict__ w, _Float16* __restrict__ A,
                            int M, int Kreal, int Kpad) {
  int idx = blockIdx.x * 256 + threadIdx.x;
  if (idx >= M * Kpad) return;
  int m = idx / Kpad, k = idx - m * Kpad;
  float v = (k < Kreal) ? w[(size_t)m * Kreal + k] : 0.0f;
  A[idx] = (_Float16)v;
}

/* ---------------- im2col: X [Cin][NN] fp32 -> Bm [Kpad][NN] f16 (streaming) ------- */
/* k = ci*25 + kh*5 + kw ; SAME pad 2 ; rows >= Cin*25 zero-filled */
__global__ void k_im2col(const float* __restrict__ X, _Float16* __restrict__ Bm, int Cin) {
  int k = blockIdx.y;
  int n = blockIdx.x * 256 + threadIdx.x;
  float v = 0.0f;
  int Kreal = Cin * 25;
  if (k < Kreal) {
    int ci = k / 25, r = k - ci * 25, kh = r / 5, kw = r - kh * 5;
    int b = n >> 10, p = n & 1023, h = p >> 5, w = p & 31;
    int hh = h + kh - 2, ww = w + kw - 2;
    if (hh >= 0 && hh < 32 && ww >= 0 && ww < 32)
      v = X[(size_t)ci * NN + (b << 10) + (hh << 5) + ww];
  }
  __builtin_nontemporal_store((_Float16)v, &Bm[(size_t)k * NN + n]);
}

/* ---------------- WMMA GEMM: C[M][NN] = A[M][K] * Bm[K][NN] + bias ----------------
   block: 256 thr = 8 wave32; tile 64(M) x 256(N); K chunks of 32.
   A staged row-major [64][32]; B staged TRANSPOSED [256 cols][32 k] so each
   lane's fragment = two contiguous 8-half runs -> ds_load_b128.
   B fragments double-buffered to overlap DS latency with WMMA.                */
__global__ void __launch_bounds__(256) k_gemm_wmma(
    const _Float16* __restrict__ A, const _Float16* __restrict__ Bm,
    const float* __restrict__ bias, float* __restrict__ C, int M, int K) {
  __shared__ __align__(16) _Float16 As[64][40];    /* [row][k], padded */
  __shared__ __align__(16) _Float16 Bt[256][40];   /* [col][k], padded */
  const int tid  = threadIdx.x;
  const int lane = tid & 31, wave = tid >> 5;
  const int hl = lane & 15, hi = lane >> 4;
  const int wm = wave >> 1, wn = wave & 1;
  const int m_base = blockIdx.y * 64, n_base = blockIdx.x * 256;

  const v8f vzero = {0.f, 0.f, 0.f, 0.f, 0.f, 0.f, 0.f, 0.f};
  v8f acc[8] = {vzero, vzero, vzero, vzero, vzero, vzero, vzero, vzero};

  const int arow = tid >> 2, acol = (tid & 3) << 3;   /* A stage: 8 halfs/thr   */
  const int bq   = tid >> 5, bg   = tid & 31;         /* B stage: 4k x 8col/thr */
  const int nk = K >> 5;

  auto loadB = [&](v16h& bf, int j) {
    const int col = (wn << 7) + (j << 4) + hl;
    const _Float16* bp = &Bt[col][0];
    float4* bfp = reinterpret_cast<float4*>(&bf);
    bfp[0] = *reinterpret_cast<const float4*>(bp + (hi << 3));
    bfp[1] = *reinterpret_cast<const float4*>(bp + 16 + (hi << 3));
  };

  for (int kc = 0; kc < nk; ++kc) {
    const int k0 = kc << 5;
    /* ---- stage A tile 64x32 ---- */
    {
      int gm = m_base + arow;
      float4 va = make_float4(0.f, 0.f, 0.f, 0.f);
      if (gm < M) va = *reinterpret_cast<const float4*>(A + (size_t)gm * K + k0 + acol);
      *reinterpret_cast<float4*>(&As[arow][acol]) = va;
    }
    /* ---- stage B tile 32x256, transposed into Bt[col][k]; streaming loads ---- */
    {
      const _Float16* bp = Bm + (size_t)(k0 + 4 * bq) * NN + n_base + (bg << 3);
      u32x4 r0 = __builtin_nontemporal_load(reinterpret_cast<const u32x4*>(bp));
      u32x4 r1 = __builtin_nontemporal_load(reinterpret_cast<const u32x4*>(bp + NN));
      u32x4 r2 = __builtin_nontemporal_load(reinterpret_cast<const u32x4*>(bp + 2 * NN));
      u32x4 r3 = __builtin_nontemporal_load(reinterpret_cast<const u32x4*>(bp + 3 * NN));
      const _Float16* h0 = reinterpret_cast<const _Float16*>(&r0);
      const _Float16* h1 = reinterpret_cast<const _Float16*>(&r1);
      const _Float16* h2 = reinterpret_cast<const _Float16*>(&r2);
      const _Float16* h3 = reinterpret_cast<const _Float16*>(&r3);
      #pragma unroll
      for (int e = 0; e < 8; ++e) {
        union { uint2 u2; _Float16 h[4]; } pk;
        pk.h[0] = h0[e]; pk.h[1] = h1[e]; pk.h[2] = h2[e]; pk.h[3] = h3[e];
        *reinterpret_cast<uint2*>(&Bt[(bg << 3) + e][bq << 2]) = pk.u2;
      }
      if (kc + 1 < nk)
        __builtin_prefetch(Bm + (size_t)(k0 + 32 + 4 * bq) * NN + n_base + (bg << 3), 0, 0);
    }
    __syncthreads();

    /* A fragment: lane (hl,hi) = row wm*16+hl, K runs [hi*8..+7] and [16+hi*8..+7] */
    v16h af;
    {
      const _Float16* ap = &As[(wm << 4) + hl][0];
      float4* afp = reinterpret_cast<float4*>(&af);
      afp[0] = *reinterpret_cast<const float4*>(ap + (hi << 3));
      afp[1] = *reinterpret_cast<const float4*>(ap + 16 + (hi << 3));
    }
    /* software-pipelined B fragments: load j+1 while WMMA j executes */
    v16h bf0, bf1;
    loadB(bf0, 0);
    #pragma unroll
    for (int j = 0; j < 8; ++j) {
      v16h& cur = (j & 1) ? bf1 : bf0;
      v16h& nxt = (j & 1) ? bf0 : bf1;
      if (j + 1 < 8) loadB(nxt, j + 1);
      acc[j] = __builtin_amdgcn_wmma_f32_16x16x32_f16(false, af, false, cur,
                                                      (short)0, acc[j], false, false);
    }
    __syncthreads();
  }

  /* C layout: VGPR r -> M = r + hi*8, N = hl */
  #pragma unroll
  for (int j = 0; j < 8; ++j) {
    const int gcol = n_base + (wn << 7) + (j << 4) + hl;
    #pragma unroll
    for (int r = 0; r < 8; ++r) {
      const int grow = m_base + (wm << 4) + (hi << 3) + r;
      if (grow < M) {
        float v = acc[j][r];
        if (bias) v += bias[grow];
        C[(size_t)grow * NN + gcol] = v;
      }
    }
  }
}

/* ---------------- attention: scores[b][ts] = <c_cur, hist[ts]> / 256 ---------------- */
__global__ void k_attn_scores(const float* __restrict__ histBase, int curSlot,
                              float* __restrict__ scores) {
  const int ts = blockIdx.x, b = blockIdx.y;
  const size_t SL = (size_t)HCH * NN;
  const float* cc = histBase + (size_t)curSlot * SL;
  const float* hh = histBase + (size_t)ts * SL;
  __shared__ float red[256];
  float s = 0.f;
  for (int i = threadIdx.x; i < HCH * 1024; i += 256) {
    int c = i >> 10, p = i & 1023;
    size_t o = (size_t)c * NN + (b << 10) + p;
    s += cc[o] * hh[o];
  }
  red[threadIdx.x] = s;
  __syncthreads();
  for (int st = 128; st > 0; st >>= 1) {
    if (threadIdx.x < st) red[threadIdx.x] += red[threadIdx.x + st];
    __syncthreads();
  }
  if (threadIdx.x == 0) scores[b * 32 + ts] = red[0] * (1.0f / 256.0f);
}

__global__ void k_attn_softmax(const float* __restrict__ scores, float* __restrict__ attn, int T) {
  int b = threadIdx.x;
  if (b >= BB) return;
  float mx = -1e30f;
  for (int t = 0; t < T; ++t) mx = fmaxf(mx, scores[b * 32 + t]);
  float sum = 0.f;
  for (int t = 0; t < T; ++t) sum += expf(scores[b * 32 + t] - mx);
  float inv = 1.0f / sum;
  for (int t = 0; t < T; ++t) attn[b * 32 + t] = expf(scores[b * 32 + t] - mx) * inv;
}

__global__ void k_attn_apply(const float* __restrict__ histBase, const float* __restrict__ attn,
                             float* __restrict__ catt, int T) {
  size_t idx = (size_t)blockIdx.x * 256 + threadIdx.x;   /* c*NN + n */
  int n = (int)(idx & (NN - 1));
  int b = n >> 10;
  const size_t SL = (size_t)HCH * NN;
  float s = 0.f;
  for (int t = 0; t < T; ++t) s += attn[b * 32 + t] * histBase[(size_t)t * SL + idx];
  catt[idx] = s;
}

/* ---------------- gate update: c_new, m_new, mem(=concat), f16 copy ---------------- */
__global__ void k_cell_update(const float* __restrict__ xs, const float* __restrict__ hs,
                              const float* __restrict__ ms, const float* __restrict__ catt,
                              float* __restrict__ m, float* __restrict__ cnew,
                              float* __restrict__ memb, _Float16* __restrict__ memf16) {
  size_t idx = (size_t)blockIdx.x * 256 + threadIdx.x;   /* c*NN + n, c<64 */
  const size_t R = (size_t)HCH * NN;
  float it = sigmoidf_(xs[idx] + hs[idx]);
  float ft = sigmoidf_(xs[R + idx] + hs[R + idx] + 1.0f);
  float gt = tanhf(xs[2 * R + idx] + hs[2 * R + idx]);
  float cn = ft * catt[idx] + it * gt;
  float ip = sigmoidf_(xs[3 * R + idx] + ms[idx]);
  float fp = sigmoidf_(xs[4 * R + idx] + ms[R + idx] + 1.0f);
  float gp = tanhf(xs[5 * R + idx] + ms[2 * R + idx]);
  float mn = fp * m[idx] + ip * gp;
  m[idx] = mn;
  cnew[idx] = cn;
  memb[idx] = cn;
  memb[R + idx] = mn;
  memf16[idx] = (_Float16)cn;
  memf16[R + idx] = (_Float16)mn;
}

__global__ void k_cell_output(const float* __restrict__ xs, const float* __restrict__ hs,
                              const float* __restrict__ oconv, const float* __restrict__ lastconv,
                              float* __restrict__ h, _Float16* __restrict__ hf16) {
  size_t idx = (size_t)blockIdx.x * 256 + threadIdx.x;
  const size_t R = (size_t)HCH * NN;
  float ot = sigmoidf_(xs[6 * R + idx] + hs[3 * R + idx] + oconv[idx]);
  float hn = ot * tanhf(lastconv[idx]);
  h[idx] = hn;
  hf16[idx] = (_Float16)hn;
}

/* ---------------- input prep (scheduled sampling) & output writer ---------------- */
__global__ void k_prep_x(const float* __restrict__ frames, const float* __restrict__ mask,
                         const float* __restrict__ xgen, float* __restrict__ xcur, int t) {
  size_t idx = (size_t)blockIdx.x * 256 + threadIdx.x;   /* c*NN + n, c<16 */
  int c = (int)(idx >> 14);
  int n = (int)(idx & (NN - 1));
  int b = n >> 10, p = n & 1023, h = p >> 5, w = p & 31;
  size_t fidx = ((((size_t)b * TT + t) * HWD + h) * HWD + w) * FCH + c;
  float fr = frames[fidx];
  float v;
  if (t < TIN) {
    v = fr;
  } else {
    size_t midx = ((((size_t)b * (TT - TIN - 1) + (t - TIN)) * HWD + h) * HWD + w) * FCH + c;
    float mv = mask[midx];
    v = mv * fr + (1.0f - mv) * xgen[(size_t)c * NN + n];
  }
  xcur[idx] = v;
}

__global__ void k_write_out(const float* __restrict__ xgen, float* __restrict__ out, int t) {
  size_t idx = (size_t)blockIdx.x * 256 + threadIdx.x;   /* c*NN + n */
  int c = (int)(idx >> 14);
  int n = (int)(idx & (NN - 1));
  int b = n >> 10, p = n & 1023, h = p >> 5, w = p & 31;
  out[((((size_t)b * NSTEP + t) * HWD + h) * HWD + w) * FCH + c] = xgen[idx];
}

extern "C" void kernel_launch(void* const* d_in, const int* in_sizes, int n_in,
                              void* d_out, int out_size, void* d_ws, size_t ws_size,
                              hipStream_t stream) {
  (void)in_sizes; (void)n_in; (void)out_size;
  const float* frames = (const float*)d_in[0];
  const float* mask   = (const float*)d_in[1];
  const float* conv_last_w = (const float*)d_in[2 + NLAY * 10];
  auto P = [&](int l, int j) -> const float* { return (const float*)d_in[2 + l * 10 + j]; };

  char* ws = (char*)d_ws;
  size_t off = 0;
  auto allocF = [&](size_t elems) -> float* {
    off = (off + 255) & ~(size_t)255;
    float* p = (float*)(ws + off); off += elems * sizeof(float); return p;
  };
  auto allocH = [&](size_t elems) -> _Float16* {
    off = (off + 255) & ~(size_t)255;
    _Float16* p = (_Float16*)(ws + off); off += elems * sizeof(_Float16); return p;
  };

  const size_t R = (size_t)HCH * NN;   /* one 64-ch activation plane */
  float* hist = allocF((size_t)NLAY * 20 * R);
  float* hbuf[NLAY];
  for (int l = 0; l < NLAY; ++l) hbuf[l] = allocF(R);
  float* mbuf   = allocF(R);
  float* xs     = allocF(7 * R);
  float* hsB    = allocF(4 * R);
  float* msB    = allocF(3 * R);
  float* memb   = allocF(2 * R);
  float* oconv  = allocF(R);
  float* lastc  = allocF(R);
  float* catt   = allocF(R);
  float* xcur   = allocF((size_t)FCH * NN);
  float* xgen   = allocF((size_t)FCH * NN);
  float* scores = allocF(16 * 32);
  float* attn   = allocF(16 * 32);
  _Float16* Bm     = allocH((size_t)3200 * NN);
  _Float16* memf16 = allocH(2 * R);
  _Float16* hf16   = allocH(R);
  _Float16 *wxf[NLAY], *whf[NLAY], *wmf[NLAY], *wof[NLAY], *wlf[NLAY];
  int Kxp[NLAY];
  for (int l = 0; l < NLAY; ++l) {
    int cin = l ? HCH : FCH;
    int kx = cin * 25;
    Kxp[l] = (kx + 31) & ~31;
    wxf[l] = allocH((size_t)448 * Kxp[l]);
    whf[l] = allocH((size_t)256 * 1600);
    wmf[l] = allocH((size_t)192 * 1600);
    wof[l] = allocH((size_t)64 * 3200);
    wlf[l] = allocH((size_t)64 * 128);
  }
  _Float16* clf = allocH(16 * 64);
  if (off > ws_size) return;   /* scratch too small: bail (deterministic) */

  auto convw = [&](const float* w, _Float16* A, int M, int Kreal, int Kp) {
    int n = M * Kp;
    k_convert_w<<<(n + 255) / 256, 256, 0, stream>>>(w, A, M, Kreal, Kp);
  };
  for (int l = 0; l < NLAY; ++l) {
    int cin = l ? HCH : FCH;
    convw(P(l, 0), wxf[l], 448, cin * 25, Kxp[l]);
    convw(P(l, 1), whf[l], 256, 1600, 1600);
    convw(P(l, 2), wmf[l], 192, 1600, 1600);
    convw(P(l, 3), wof[l], 64, 3200, 3200);
    convw(P(l, 4), wlf[l], 64, 128, 128);
  }
  convw(conv_last_w, clf, 16, 64, 64);

  for (int l = 0; l < NLAY; ++l) {
    hipMemsetAsync(hbuf[l], 0, R * sizeof(float), stream);
    hipMemsetAsync(hist + (size_t)l * 20 * R, 0, R * sizeof(float), stream);  /* slot 0 */
  }
  hipMemsetAsync(mbuf, 0, R * sizeof(float), stream);

  for (int t = 0; t < NSTEP; ++t) {
    k_prep_x<<<(FCH * NN) / 256, 256, 0, stream>>>(frames, mask, xgen, xcur, t);
    for (int l = 0; l < NLAY; ++l) {
      const float* xin = (l == 0) ? xcur : hbuf[l - 1];
      int cin = l ? HCH : FCH;
      float* histL = hist + (size_t)l * 20 * R;

      /* xs = conv(x, wx)+bx : M=448, K=Kxp */
      k_im2col<<<dim3(NN / 256, Kxp[l]), 256, 0, stream>>>(xin, Bm, cin);
      k_gemm_wmma<<<dim3(64, 7), 256, 0, stream>>>(wxf[l], Bm, P(l, 5), xs, 448, Kxp[l]);
      /* hs = conv(h, wh)+bh : M=256 K=1600 */
      k_im2col<<<dim3(NN / 256, 1600), 256, 0, stream>>>(hbuf[l], Bm, HCH);
      k_gemm_wmma<<<dim3(64, 4), 256, 0, stream>>>(whf[l], Bm, P(l, 6), hsB, 256, 1600);
      /* ms = conv(m, wm)+bm : M=192 K=1600 */
      k_im2col<<<dim3(NN / 256, 1600), 256, 0, stream>>>(mbuf, Bm, HCH);
      k_gemm_wmma<<<dim3(64, 3), 256, 0, stream>>>(wmf[l], Bm, P(l, 7), msB, 192, 1600);

      /* temporal recall attention over history slots [0..t], c_cur = slot t */
      int T = t + 1;
      k_attn_scores<<<dim3(T, BB), 256, 0, stream>>>(histL, t, scores);
      k_attn_softmax<<<1, 32, 0, stream>>>(scores, attn, T);
      k_attn_apply<<<(HCH * NN) / 256, 256, 0, stream>>>(histL, attn, catt, T);

      /* gates; c_new appended at slot t+1 */
      k_cell_update<<<(HCH * NN) / 256, 256, 0, stream>>>(
          xs, hsB, msB, catt, mbuf, histL + (size_t)(t + 1) * R, memb, memf16);

      /* o-gate conv on mem (128ch) + 1x1 last conv */
      k_im2col<<<dim3(NN / 256, 3200), 256, 0, stream>>>(memb, Bm, 2 * HCH);
      k_gemm_wmma<<<dim3(64, 1), 256, 0, stream>>>(wof[l], Bm, P(l, 8), oconv, 64, 3200);
      k_gemm_wmma<<<dim3(64, 1), 256, 0, stream>>>(wlf[l], memf16, P(l, 9), lastc, 64, 128);
      k_cell_output<<<(HCH * NN) / 256, 256, 0, stream>>>(xs, hsB, oconv, lastc, hbuf[l], hf16);
    }
    /* x_gen = 1x1 conv(h[3]) : M=16 K=64 */
    k_gemm_wmma<<<dim3(64, 1), 256, 0, stream>>>(clf, hf16, (const float*)nullptr, xgen, FCH, 64);
    k_write_out<<<(FCH * NN) / 256, 256, 0, stream>>>(xgen, (float*)d_out, t);
  }
}
